// TransformerGenerator_62397284876459
// MI455X (gfx1250) — matrix operand order repaired and checked by reading, not compiled, not run on hardware
//
#include <hip/hip_runtime.h>
#include <math.h>
#include <stdint.h>

// ---------------------------------------------------------------------------
// Problem constants (from reference)
// ---------------------------------------------------------------------------
#define NL 6       // layers
#define HD 512     // hidden dim
#define SEQ 2048   // sequence length
#define NB 8       // batch
#define FF 2048    // ffn dim = 4*HD

// ---------------------------------------------------------------------------
// Vector types for WMMA (wave32, gfx1250)
// ---------------------------------------------------------------------------
typedef __attribute__((ext_vector_type(16))) __bf16 bf16x16;
typedef __attribute__((ext_vector_type(8)))  __bf16 bf16x8;
typedef __attribute__((ext_vector_type(8)))  float  f32x8;

__device__ __forceinline__ float wave_sum(float v) {
#pragma unroll
  for (int off = 16; off > 0; off >>= 1) v += __shfl_xor(v, off, 32);
  return v;
}
__device__ __forceinline__ float wave_max(float v) {
#pragma unroll
  for (int off = 16; off > 0; off >>= 1) v = fmaxf(v, __shfl_xor(v, off, 32));
  return v;
}
__device__ __forceinline__ bf16x16 cat8(bf16x8 lo, bf16x8 hi) {
  return __builtin_shufflevector(lo, hi, 0,1,2,3,4,5,6,7,8,9,10,11,12,13,14,15);
}

// Async global->LDS copy of 16B per lane (GLOBAL_LOAD_ASYNC_TO_LDS_B128,
// tracked by ASYNCcnt). LDS dest address = low 32 bits of the flat shared
// pointer (LDS aperture truncation, ISA 10.2).
__device__ __forceinline__ void async_ld16(uint32_t lds, const void* g) {
  asm volatile("global_load_async_to_lds_b128 %0, %1, off"
               :: "v"(lds), "v"((unsigned long long)(uintptr_t)g)
               : "memory");
}
__device__ __forceinline__ void wait_async0() {
  asm volatile("s_wait_asynccnt 0x0" ::: "memory");
}

// ---------------------------------------------------------------------------
// Tiled NT GEMM:  C[M,N] = epilogue( scale * A[M,K] * B[N,K]^T )
// A, B bf16 row-major with K contiguous. blockIdx.z = batch (stride params).
// Block tile 128x256, 8 waves in a 2(m) x 4(n) grid; wave tile 64x64:
// 16x v_wmma_f32_16x16x32_bf16 per K-step of 32. Tiles are brought into LDS
// with async global->LDS (double-buffered, one barrier per K-step).
// M % 128 == 0, N % 256 == 0, K % 32 == 0 (true for all uses here).
// ---------------------------------------------------------------------------
#define BM 128
#define BN 256
#define BK 32
#define LDSS 40   // BK + 8 bf16 padding -> conflict-free ds_load_b128

template<bool BIAS, bool RELU, bool RESID, bool OUT_BF16>
__global__ __launch_bounds__(256)
void gemm_nt(const __bf16* __restrict__ A, long batchA,
             const __bf16* __restrict__ B, long batchB,
             const float*  __restrict__ bias,
             const float*  __restrict__ resid,
             void* __restrict__ Cout, long batchC,
             int N, int K, float scale)
{
  __shared__ __bf16 As[2][BM * LDSS];
  __shared__ __bf16 Bs[2][BN * LDSS];

  const int tid  = threadIdx.x;
  const int lane = tid & 31;
  const int wave = tid >> 5;
  const int wm   = wave >> 2;          // 0..1
  const int wn   = wave & 3;           // 0..3
  const int bm   = blockIdx.y * BM;
  const int bn   = blockIdx.x * BN;
  const long bz  = blockIdx.z;

  // async-copy mapping: A tile = 128 rows x 64B (2 x 16B slots/thread),
  //                     B tile = 256 rows x 64B (4 x 16B slots/thread)
  const __bf16* Ag = A + bz * batchA + (size_t)(bm + (tid >> 1)) * K + (tid & 1) * 16;
  const __bf16* Bg = B + bz * batchB + (size_t)(bn + tid) * K;
  const uint32_t aL0 = (uint32_t)(uintptr_t)(&As[0][0])
                     + (uint32_t)(tid >> 1) * (LDSS * 2) + (uint32_t)(tid & 1) * 32;
  const uint32_t bL0 = (uint32_t)(uintptr_t)(&Bs[0][0])
                     + (uint32_t)tid * (LDSS * 2);
  const uint32_t abuf = BM * LDSS * 2;   // bytes between As[0] and As[1]
  const uint32_t bbuf = BN * LDSS * 2;

  auto issue_tile = [&](int kt, int buf) {
    const __bf16* ga = Ag + kt * BK;
    const __bf16* gb = Bg + kt * BK;
    const uint32_t al = aL0 + (uint32_t)buf * abuf;
    const uint32_t bl = bL0 + (uint32_t)buf * bbuf;
    async_ld16(al,      ga);
    async_ld16(al + 16, ga + 8);
    async_ld16(bl,      gb);
    async_ld16(bl + 16, gb + 8);
    async_ld16(bl + 32, gb + 16);
    async_ld16(bl + 48, gb + 24);
  };

  f32x8 acc[4][4];
#pragma unroll
  for (int i = 0; i < 4; ++i)
#pragma unroll
    for (int j = 0; j < 4; ++j)
#pragma unroll
      for (int e = 0; e < 8; ++e) acc[i][j][e] = 0.0f;

  // LDS fragment addressing (ISA 7.12.2 bf16 layouts):
  //  A 16x32: lane<16 holds row=lane, K{0..7,16..23}; lane>=16 row=lane-16, K{8..15,24..31}
  //  B 32x16: lane<16 holds col=lane, K{0..7,16..23};  lane>=16 col=lane-16, K{8..15,24..31}
  const int arow = wm * 64 + (lane & 15);
  const int akb  = (lane >> 4) * 8;
  const int bcol = wn * 64 + (lane & 15);
  const int bkb  = (lane >> 4) * 8;

  const int KT = K / BK;

  issue_tile(0, 0);
  wait_async0();
  __syncthreads();

  for (int kt = 0; kt < KT; ++kt) {
    const int p = kt & 1;
    if (kt + 1 < KT) issue_tile(kt + 1, p ^ 1);   // fill other buffer async

    bf16x16 fa[4], fb[4];
#pragma unroll
    for (int i = 0; i < 4; ++i) {
      const __bf16* q = &As[p][(arow + i * 16) * LDSS + akb];
      fa[i] = cat8(*(const bf16x8*)q, *(const bf16x8*)(q + 16));
    }
#pragma unroll
    for (int j = 0; j < 4; ++j) {
      const __bf16* q = &Bs[p][(bcol + j * 16) * LDSS + bkb];
      fb[j] = cat8(*(const bf16x8*)q, *(const bf16x8*)(q + 16));
    }
#pragma unroll
    for (int i = 0; i < 4; ++i)
#pragma unroll
      for (int j = 0; j < 4; ++j)
        acc[i][j] = __builtin_amdgcn_wmma_f32_16x16x32_bf16(
            false, fa[i], false, fb[j], (short)0, acc[i][j], false, false);

    wait_async0();     // my slice of next tile has landed in LDS
    __syncthreads();   // everyone's slice has landed; buf p reads done
  }

  // C layout (ISA 7.12.2): VGPR r, lane l -> row = r + 8*(l>=16), col = l%16
  const int row0 = bm + wm * 64 + ((lane >> 4) << 3);
  const int col0 = bn + wn * 64 + (lane & 15);
#pragma unroll
  for (int i = 0; i < 4; ++i)
#pragma unroll
    for (int j = 0; j < 4; ++j) {
      const int col = col0 + j * 16;
      const float bv = BIAS ? bias[col] : 0.0f;
#pragma unroll
      for (int r = 0; r < 8; ++r) {
        const int row = row0 + i * 16 + r;
        float v = acc[i][j][r] * scale + bv;
        if (RELU) v = fmaxf(v, 0.0f);
        if (RESID) v += resid[(size_t)row * N + col];
        const size_t off = (size_t)bz * batchC + (size_t)row * N + col;
        if (OUT_BF16) ((__bf16*)Cout)[off] = (__bf16)v;
        else          ((float*)Cout)[off]  = v;
      }
    }
}

// ---------------------------------------------------------------------------
// Encoder: x[b,s,d] = z*enc_w[d,0] + c*enc_w[d,1] + enc_b[d] + posenc(s,d)
// ---------------------------------------------------------------------------
__global__ __launch_bounds__(256)
void encode_kernel(const float* __restrict__ z, const float* __restrict__ c,
                   const float* __restrict__ ew, const float* __restrict__ eb,
                   float* __restrict__ x)
{
  const long idx = (long)blockIdx.x * 256 + threadIdx.x;
  const int  d   = (int)(idx & (HD - 1));
  const long bs  = idx >> 9;                // /HD
  const int  s   = (int)(bs & (SEQ - 1));
  // div = exp(-(ln 10000)/D * (d & ~1))
  const float ang = (float)s * __expf(-0.017988946039015982f * (float)(d & ~1));
  const float pe  = (d & 1) ? __cosf(ang) : __sinf(ang);
  x[idx] = z[bs] * ew[2 * d] + c[bs] * ew[2 * d + 1] + eb[d] + pe;
}

// ---------------------------------------------------------------------------
// f32 -> bf16 weight conversion
// ---------------------------------------------------------------------------
__global__ __launch_bounds__(256)
void f32_to_bf16(const float* __restrict__ in, __bf16* __restrict__ out, long n)
{
  const long i = (long)blockIdx.x * 256 + threadIdx.x;
  if (i < n) out[i] = (__bf16)in[i];
}

// ---------------------------------------------------------------------------
// LayerNorm (wave per row, D=512 -> 16 elems/lane); bf16 out
// ---------------------------------------------------------------------------
__global__ __launch_bounds__(256)
void ln_kernel(const float* __restrict__ x, const float* __restrict__ w,
               const float* __restrict__ b, __bf16* __restrict__ out)
{
  const int lane = threadIdx.x & 31;
  const int row  = blockIdx.x * 8 + (threadIdx.x >> 5);
  const size_t base = (size_t)row * HD;
  float v[16];
  float s = 0.0f;
#pragma unroll
  for (int i = 0; i < 16; ++i) { v[i] = x[base + i * 32 + lane]; s += v[i]; }
  s = wave_sum(s);
  const float mean = s * (1.0f / HD);
  float q = 0.0f;
#pragma unroll
  for (int i = 0; i < 16; ++i) { const float d = v[i] - mean; q += d * d; }
  q = wave_sum(q);
  const float rstd = rsqrtf(q * (1.0f / HD) + 1e-5f);
#pragma unroll
  for (int i = 0; i < 16; ++i) {
    const int d = i * 32 + lane;
    out[base + d] = (__bf16)((v[i] - mean) * rstd * w[d] + b[d]);
  }
}

// ---------------------------------------------------------------------------
// Fused: x = LN1(x + atto)  (written back f32)  then  out2 = LN2(x)  (bf16)
// ---------------------------------------------------------------------------
__global__ __launch_bounds__(256)
void ln_fused(float* __restrict__ x, const float* __restrict__ atto,
              const float* __restrict__ w1, const float* __restrict__ b1,
              const float* __restrict__ w2, const float* __restrict__ b2,
              __bf16* __restrict__ out2)
{
  const int lane = threadIdx.x & 31;
  const int row  = blockIdx.x * 8 + (threadIdx.x >> 5);
  const size_t base = (size_t)row * HD;
  float v[16];
  float s = 0.0f;
#pragma unroll
  for (int i = 0; i < 16; ++i) {
    const int d = i * 32 + lane;
    v[i] = x[base + d] + atto[base + d];
    s += v[i];
  }
  s = wave_sum(s);
  const float mean = s * (1.0f / HD);
  float q = 0.0f;
#pragma unroll
  for (int i = 0; i < 16; ++i) { const float d = v[i] - mean; q += d * d; }
  q = wave_sum(q);
  const float rstd = rsqrtf(q * (1.0f / HD) + 1e-5f);
  float y[16];
  float s2 = 0.0f;
#pragma unroll
  for (int i = 0; i < 16; ++i) {
    const int d = i * 32 + lane;
    y[i] = (v[i] - mean) * rstd * w1[d] + b1[d];
    x[base + d] = y[i];
    s2 += y[i];
  }
  s2 = wave_sum(s2);
  const float mean2 = s2 * (1.0f / HD);
  float q2 = 0.0f;
#pragma unroll
  for (int i = 0; i < 16; ++i) { const float d = y[i] - mean2; q2 += d * d; }
  q2 = wave_sum(q2);
  const float rstd2 = rsqrtf(q2 * (1.0f / HD) + 1e-5f);
#pragma unroll
  for (int i = 0; i < 16; ++i) {
    const int d = i * 32 + lane;
    out2[base + d] = (__bf16)((y[i] - mean2) * rstd2 * w2[d] + b2[d]);
  }
}

// ---------------------------------------------------------------------------
// Softmax over S with cosine positional bias computed on the fly.
// probs[b,i,:] = softmax(scores[b,i,:] + b0*cos(2pi*d/24) + b1*cos(2pi*d/720))
// ---------------------------------------------------------------------------
__global__ __launch_bounds__(256)
void softmax_bias(const float* __restrict__ scores, __bf16* __restrict__ probs,
                  const float* __restrict__ beta, int layer)
{
  __shared__ float red[8];
  const int i = blockIdx.x;
  const int b = blockIdx.y;
  const float b0 = beta[layer * 2 + 0];
  const float b1 = beta[layer * 2 + 1];
  const size_t base = ((size_t)b * SEQ + i) * SEQ;

  float vals[8];
  float mx = -3.0e38f;
#pragma unroll
  for (int t = 0; t < 8; ++t) {
    const int j = t * 256 + threadIdx.x;
    const float dist = fabsf((float)(i - j));
    const float bias = b0 * __cosf(0.26179938779914946f * dist)   // 2*pi/24
                     + b1 * __cosf(0.008726646259971648f * dist); // 2*pi/720
    vals[t] = scores[base + j] + bias;
    mx = fmaxf(mx, vals[t]);
  }
  mx = wave_max(mx);
  if ((threadIdx.x & 31) == 0) red[threadIdx.x >> 5] = mx;
  __syncthreads();
  mx = red[0];
#pragma unroll
  for (int w = 1; w < 8; ++w) mx = fmaxf(mx, red[w]);
  __syncthreads();

  float sum = 0.0f;
#pragma unroll
  for (int t = 0; t < 8; ++t) { vals[t] = __expf(vals[t] - mx); sum += vals[t]; }
  sum = wave_sum(sum);
  if ((threadIdx.x & 31) == 0) red[threadIdx.x >> 5] = sum;
  __syncthreads();
  sum = 0.0f;
#pragma unroll
  for (int w = 0; w < 8; ++w) sum += red[w];
  const float inv = 1.0f / sum;
#pragma unroll
  for (int t = 0; t < 8; ++t) {
    const int j = t * 256 + threadIdx.x;
    probs[base + j] = (__bf16)(vals[t] * inv);
  }
}

// ---------------------------------------------------------------------------
// Per-batch transpose: Vt[b][d][s] = V[b][s][d]  (bf16, 32x32 LDS tiles)
// ---------------------------------------------------------------------------
__global__ __launch_bounds__(256)
void transpose_bf16(const __bf16* __restrict__ V, __bf16* __restrict__ Vt)
{
  __shared__ __bf16 t[32][33];
  const int b  = blockIdx.z;
  const int s0 = blockIdx.x * 32;
  const int d0 = blockIdx.y * 32;
  const __bf16* Vb  = V  + (size_t)b * SEQ * HD;
  __bf16*       Vtb = Vt + (size_t)b * SEQ * HD;
#pragma unroll
  for (int r = 0; r < 32; r += 8)
    t[threadIdx.y + r][threadIdx.x] =
        Vb[(size_t)(s0 + threadIdx.y + r) * HD + d0 + threadIdx.x];
  __syncthreads();
#pragma unroll
  for (int r = 0; r < 32; r += 8)
    Vtb[(size_t)(d0 + threadIdx.y + r) * SEQ + s0 + threadIdx.x] =
        t[threadIdx.x][threadIdx.y + r];
}

// ---------------------------------------------------------------------------
// Output head: out[row] = dot(x[row,:], out_w) + out_b   (wave per row)
// ---------------------------------------------------------------------------
__global__ __launch_bounds__(256)
void out_proj(const float* __restrict__ x, const float* __restrict__ ow,
              const float* __restrict__ ob, float* __restrict__ out)
{
  const int lane = threadIdx.x & 31;
  const int row  = blockIdx.x * 8 + (threadIdx.x >> 5);
  const size_t base = (size_t)row * HD;
  float s = 0.0f;
#pragma unroll
  for (int i = 0; i < 16; ++i) s += x[base + i * 32 + lane] * ow[i * 32 + lane];
  s = wave_sum(s);
  if (lane == 0) out[row] = s + ob[0];
}

// ---------------------------------------------------------------------------
// Host orchestration
// ---------------------------------------------------------------------------
extern "C" void kernel_launch(void* const* d_in, const int* in_sizes, int n_in,
                              void* d_out, int out_size, void* d_ws, size_t ws_size,
                              hipStream_t stream)
{
  (void)in_sizes; (void)n_in; (void)out_size; (void)ws_size;

  const float* z    = (const float*)d_in[0];
  const float* c    = (const float*)d_in[1];
  const float* encw = (const float*)d_in[2];
  const float* encb = (const float*)d_in[3];
  const float* beta = (const float*)d_in[4];
  const float* wq   = (const float*)d_in[5];
  const float* bq   = (const float*)d_in[6];
  const float* wk   = (const float*)d_in[7];
  const float* bk   = (const float*)d_in[8];
  const float* wv   = (const float*)d_in[9];
  const float* bv   = (const float*)d_in[10];
  const float* ln1w = (const float*)d_in[11];
  const float* ln1b = (const float*)d_in[12];
  const float* ln2w = (const float*)d_in[13];
  const float* ln2b = (const float*)d_in[14];
  const float* f1w  = (const float*)d_in[15];
  const float* f1b  = (const float*)d_in[16];
  const float* f2w  = (const float*)d_in[17];
  const float* f2b  = (const float*)d_in[18];
  const float* ow   = (const float*)d_in[19];
  const float* ob   = (const float*)d_in[20];

  const int R = NB * SEQ;  // 16384 rows

  uintptr_t cur = (uintptr_t)d_ws;
  auto carve = [&](size_t bytes) -> void* {
    uintptr_t p = (cur + 255) & ~(uintptr_t)255;
    cur = p + bytes;
    return (void*)p;
  };

  __bf16* wqB = (__bf16*)carve((size_t)NL * HD * HD * 2);
  __bf16* wkB = (__bf16*)carve((size_t)NL * HD * HD * 2);
  __bf16* wvB = (__bf16*)carve((size_t)NL * HD * HD * 2);
  __bf16* f1B = (__bf16*)carve((size_t)NL * FF * HD * 2);
  __bf16* f2B = (__bf16*)carve((size_t)NL * HD * FF * 2);
  float*  x      = (float*) carve((size_t)R * HD * 4);
  __bf16* xn     = (__bf16*)carve((size_t)R * HD * 2);
  __bf16* qB     = (__bf16*)carve((size_t)R * HD * 2);
  __bf16* kB     = (__bf16*)carve((size_t)R * HD * 2);
  __bf16* vB     = (__bf16*)carve((size_t)R * HD * 2);
  __bf16* vtB    = (__bf16*)carve((size_t)R * HD * 2);
  float*  scores = (float*) carve((size_t)NB * SEQ * SEQ * 4);
  __bf16* probs  = (__bf16*)carve((size_t)NB * SEQ * SEQ * 2);
  float*  atto   = (float*) carve((size_t)R * HD * 4);
  __bf16* h      = (__bf16*)scores;   // reuse: scores dead once probs written

  // --- weights to bf16 (once per launch; deterministic) ---
  const long nqkv = (long)NL * HD * HD;
  const long nff  = (long)NL * FF * HD;
  f32_to_bf16<<<(unsigned)((nqkv + 255) / 256), 256, 0, stream>>>(wq, wqB, nqkv);
  f32_to_bf16<<<(unsigned)((nqkv + 255) / 256), 256, 0, stream>>>(wk, wkB, nqkv);
  f32_to_bf16<<<(unsigned)((nqkv + 255) / 256), 256, 0, stream>>>(wv, wvB, nqkv);
  f32_to_bf16<<<(unsigned)((nff  + 255) / 256), 256, 0, stream>>>(f1w, f1B, nff);
  f32_to_bf16<<<(unsigned)((nff  + 255) / 256), 256, 0, stream>>>(f2w, f2B, nff);

  // --- encoder + positional encoding ---
  encode_kernel<<<(R * HD) / 256, 256, 0, stream>>>(z, c, encw, encb, x);

  const float qscale = 1.0f / sqrtf((float)HD);   // folded into Q

  for (int l = 0; l < NL; ++l) {
    // pre-norm
    ln_kernel<<<R / 8, 256, 0, stream>>>(x, ln1w + l * HD, ln1b + l * HD, xn);

    // Q,K,V projections (bias; Q pre-scaled by 1/sqrt(D))
    dim3 gQKV(HD / BN, R / BM, 1);
    gemm_nt<true, false, false, true><<<gQKV, 256, 0, stream>>>(
        xn, 0, wqB + (size_t)l * HD * HD, 0, bq + l * HD, nullptr,
        qB, 0, HD, HD, qscale);
    gemm_nt<true, false, false, true><<<gQKV, 256, 0, stream>>>(
        xn, 0, wkB + (size_t)l * HD * HD, 0, bk + l * HD, nullptr,
        kB, 0, HD, HD, 1.0f);
    gemm_nt<true, false, false, true><<<gQKV, 256, 0, stream>>>(
        xn, 0, wvB + (size_t)l * HD * HD, 0, bv + l * HD, nullptr,
        vB, 0, HD, HD, 1.0f);

    // V -> V^T so probs@V is also an NT GEMM
    transpose_bf16<<<dim3(SEQ / 32, HD / 32, NB), dim3(32, 8), 0, stream>>>(vB, vtB);

    // scores[b] = Q[b] K[b]^T   (batched over z)
    gemm_nt<false, false, false, false><<<dim3(SEQ / BN, SEQ / BM, NB), 256, 0, stream>>>(
        qB, (long)SEQ * HD, kB, (long)SEQ * HD, nullptr, nullptr,
        scores, (long)SEQ * SEQ, SEQ, HD, 1.0f);

    // softmax with cosine positional bias
    softmax_bias<<<dim3(SEQ, NB), 256, 0, stream>>>(scores, probs, beta, l);

    // atto[b] = probs[b] @ V[b]  (= probs * Vt^T, NT)
    gemm_nt<false, false, false, false><<<dim3(HD / BN, SEQ / BM, NB), 256, 0, stream>>>(
        probs, (long)SEQ * SEQ, vtB, (long)HD * SEQ, nullptr, nullptr,
        atto, (long)SEQ * HD, HD, SEQ, 1.0f);

    // x = LN1(x + atto); xn = LN2(x)
    ln_fused<<<R / 8, 256, 0, stream>>>(x, atto, ln1w + l * HD, ln1b + l * HD,
                                        ln2w + l * HD, ln2b + l * HD, xn);

    // h = relu(xn @ ff1^T + b1)
    gemm_nt<true, true, false, true><<<dim3(FF / BN, R / BM, 1), 256, 0, stream>>>(
        xn, 0, f1B + (size_t)l * FF * HD, 0, f1b + l * FF, nullptr,
        h, 0, FF, HD, 1.0f);

    // x = x + h @ ff2^T + b2
    gemm_nt<true, false, true, false><<<dim3(HD / BN, R / BM, 1), 256, 0, stream>>>(
        h, 0, f2B + (size_t)l * HD * FF, 0, f2b + l * HD, x,
        x, 0, HD, FF, 1.0f);
  }

  // output head
  out_proj<<<R / 8, 256, 0, stream>>>(x, ow, ob, (float*)d_out);
}
